// MambaClass_63720134803773
// MI455X (gfx1250) — matrix operand order repaired and checked by reading, not compiled, not run on hardware
//
#include <hip/hip_runtime.h>
#include <hip/hip_bf16.h>

// ---------------- problem constants ----------------
#define BB      2
#define LL      5000
#define C_IN    12
#define D_MODEL 512
#define D_STATE 64
#define D_CONV  4
#define D_INNER 1024          // 2*D_MODEL / 2 halves of xz
#define DT_RANK 32
#define ML      (BB*LL)       // 10000 rows for all GEMMs

typedef __bf16 bf16_t;
typedef __attribute__((ext_vector_type(16))) __bf16 v16bf;
typedef __attribute__((ext_vector_type(8)))  float  v8f;

// ---------------- generic bf16 WMMA GEMM ----------------
// C[M,N] (f32) = A[M,K] (bf16, row-major) * B[N,K]^T (bf16, row-major)
// Double-buffered LDS staging via GLOBAL_LOAD_ASYNC_TO_LDS_B128 (ASYNCcnt),
// one workgroup barrier per K-iteration.
#define BM 128
#define BN 128
#define BK 32
#define LDS_STRIDE 40   // BK + 8 bf16 pad, keeps 16B alignment (80B rows)

__global__ __launch_bounds__(256) void gemm_bf16_wmma(
    const bf16_t* __restrict__ A, const bf16_t* __restrict__ B,
    float* __restrict__ C, int M, int N, int K)
{
    __shared__ bf16_t As[2][BM * LDS_STRIDE];
    __shared__ bf16_t Bs[2][BN * LDS_STRIDE];

    const int tid  = threadIdx.x;
    const int lane = tid & 31;
    const int wv   = tid >> 5;            // 8 waves
    const int wm   = wv >> 2;             // 2 wave-rows (64 M each)
    const int wn   = wv & 3;              // 4 wave-cols (32 N each)
    const int bm   = blockIdx.y;
    const int bn   = blockIdx.x;
    const int lrow = lane & 15;
    const int hsel = lane >> 4;           // 0: lanes 0-15, 1: lanes 16-31

    v8f acc[4][2];
#pragma unroll
    for (int i = 0; i < 4; ++i)
#pragma unroll
        for (int j = 0; j < 2; ++j)
#pragma unroll
            for (int e = 0; e < 8; ++e) acc[i][j][e] = 0.f;

    // async-stage one 128x32 A tile + 128x32 B tile into buffer `buf`
    auto stage = [&](int k0, int buf) {
#pragma unroll
        for (int c = 0; c < 2; ++c) {
            int chunk = tid + c * 256;          // 512 chunks of 8 bf16 per tile
            int row = chunk >> 2;
            int kc  = (chunk & 3) << 3;
            int gm = bm * BM + row; if (gm >= M) gm = M - 1;
            const bf16_t* ga = A + (size_t)gm * K + k0 + kc;
            unsigned la = (unsigned)(size_t)&As[buf][row * LDS_STRIDE + kc];
            asm volatile("global_load_async_to_lds_b128 %0, %1, off"
                         :: "v"(la), "v"(ga) : "memory");
            int gn = bn * BN + row; if (gn >= N) gn = N - 1;
            const bf16_t* gb = B + (size_t)gn * K + k0 + kc;
            unsigned lb = (unsigned)(size_t)&Bs[buf][row * LDS_STRIDE + kc];
            asm volatile("global_load_async_to_lds_b128 %0, %1, off"
                         :: "v"(lb), "v"(gb) : "memory");
        }
    };

    stage(0, 0);
    const int nk = K / BK;

    for (int t = 0; t < nk; ++t) {
        const int buf = t & 1;
        // my async loads for `buf` done; barrier publishes whole tile and
        // guarantees everyone finished reading the other buffer.
        asm volatile("s_wait_asynccnt 0x0" ::: "memory");
        __syncthreads();
        if (t + 1 < nk) stage((t + 1) * BK, 1 - buf);

        // A fragment (16x32 bf16): lanes 0-15 rows M, halves K {0..7,16..23};
        // lanes 16-31 same rows, K {8..15,24..31}
        v16bf afrag[4], bfrag[2];
#pragma unroll
        for (int i = 0; i < 4; ++i) {
            int row = wm * 64 + i * 16 + lrow;
            int kb  = hsel * 8;
            union { uint4 u[2]; v16bf v; } f;
            f.u[0] = *(const uint4*)&As[buf][row * LDS_STRIDE + kb];
            f.u[1] = *(const uint4*)&As[buf][row * LDS_STRIDE + kb + 16];
            afrag[i] = f.v;
        }
        // B fragment (32x16 bf16): lane = col (0-15), lanes 0-15 K {0..7,16..23},
        // lanes 16-31 K {8..15,24..31}
#pragma unroll
        for (int j = 0; j < 2; ++j) {
            int col = wn * 32 + j * 16 + lrow;
            int kb  = hsel * 8;
            union { uint4 u[2]; v16bf v; } f;
            f.u[0] = *(const uint4*)&Bs[buf][col * LDS_STRIDE + kb];
            f.u[1] = *(const uint4*)&Bs[buf][col * LDS_STRIDE + kb + 16];
            bfrag[j] = f.v;
        }

#pragma unroll
        for (int i = 0; i < 4; ++i)
#pragma unroll
            for (int j = 0; j < 2; ++j)
                acc[i][j] = __builtin_amdgcn_wmma_f32_16x16x32_bf16(
                    false, afrag[i], false, bfrag[j],
                    (short)0, acc[i][j], false, false);
    }

    // C/D layout: lanes 0-15 col=lane, vgpr v -> M=v; lanes 16-31 -> M=v+8
    const int rowbase = bm * BM + wm * 64;
    const int colbase = bn * BN + wn * 32;
#pragma unroll
    for (int i = 0; i < 4; ++i)
#pragma unroll
        for (int j = 0; j < 2; ++j) {
            int col = colbase + j * 16 + lrow;
            if (col < N) {
#pragma unroll
                for (int v = 0; v < 8; ++v) {
                    int r = rowbase + i * 16 + v + hsel * 8;
                    if (r < M) C[(size_t)r * N + col] = acc[i][j][v];
                }
            }
        }
}

// ---------------- elementwise / small kernels ----------------
__global__ void cvt_f32_bf16(const float* __restrict__ s, bf16_t* __restrict__ d, int n)
{
    int t = blockIdx.x * 256 + threadIdx.x;
    if (t < n) d[t] = (bf16_t)s[t];
}

// h[b,l,d] = sum_c x[b,c,l] * w_init[d,c]  -> bf16
__global__ void h_proj_kernel(const float* __restrict__ x, const float* __restrict__ w_init,
                              bf16_t* __restrict__ h_bf)
{
    int t = blockIdx.x * 256 + threadIdx.x;
    if (t >= ML * D_MODEL) return;
    int d  = t & (D_MODEL - 1);
    int bl = t >> 9;
    int b  = bl / LL, l = bl % LL;
    const float* xp = x + (size_t)b * C_IN * LL + l;
    const float* wp = w_init + d * C_IN;
    float acc = 0.f;
#pragma unroll
    for (int c = 0; c < C_IN; ++c) acc += xp[(size_t)c * LL] * wp[c];
    h_bf[t] = (bf16_t)acc;
}

// depthwise causal conv over xi (first half of xz) + SiLU
__global__ void conv_silu_kernel(const float* __restrict__ xz, const float* __restrict__ conv_w,
                                 const float* __restrict__ conv_b,
                                 float* __restrict__ xc, bf16_t* __restrict__ xc_bf)
{
    int t = blockIdx.x * 256 + threadIdx.x;
    if (t >= ML * D_INNER) return;
    int d  = t & (D_INNER - 1);
    int bl = t >> 10;
    int b  = bl / LL, l = bl % LL;
    float acc = conv_b[d];
    const float* w = conv_w + d * D_CONV;
#pragma unroll
    for (int k = 0; k < D_CONV; ++k) {
        int ll = l - (D_CONV - 1) + k;
        if (ll >= 0) acc += xz[((size_t)b * LL + ll) * (2 * D_INNER) + d] * w[k];
    }
    float s = acc / (1.f + __expf(-acc));
    xc[t] = s;
    xc_bf[t] = (bf16_t)s;
}

// dt_lo slice of x_dbl -> bf16 (M x 32)
__global__ void dtlo_kernel(const float* __restrict__ x_dbl, bf16_t* __restrict__ dtlo)
{
    int t = blockIdx.x * 256 + threadIdx.x;
    if (t >= ML * DT_RANK) return;
    int r = t & (DT_RANK - 1);
    int m = t >> 5;
    dtlo[t] = (bf16_t)x_dbl[(size_t)m * (DT_RANK + 2 * D_STATE) + r];
}

// dt = softplus(raw + b_dt[d])
__global__ void softplus_kernel(float* __restrict__ dt, const float* __restrict__ b_dt, int n)
{
    int t = blockIdx.x * 256 + threadIdx.x;
    if (t >= n) return;
    int d = t & (D_INNER - 1);
    float v = dt[t] + b_dt[d];
    dt[t] = (v > 20.f) ? v : log1pf(__expf(v));
}

// selective scan: one wave32 per (b,d); 2 states per lane; ys layout (b,d,l)
__global__ __launch_bounds__(256) void scan_kernel(
    const float* __restrict__ x_dbl, const float* __restrict__ dt,
    const float* __restrict__ xc, const float* __restrict__ A_log,
    float* __restrict__ ys)
{
    const int lane = threadIdx.x & 31;
    const int wv   = threadIdx.x >> 5;
    const int pair = blockIdx.x * 8 + wv;         // 0..2047 == b*1024 + d
    const int b = pair >> 10;
    const int d = pair & (D_INNER - 1);
    const int s0 = lane * 2;

    float A0 = -__expf(A_log[d * D_STATE + s0]);
    float A1 = -__expf(A_log[d * D_STATE + s0 + 1]);
    float h0 = 0.f, h1 = 0.f, yreg = 0.f;

    const float* xd  = x_dbl + (size_t)b * LL * (DT_RANK + 2 * D_STATE);
    const float* dtb = dt + (size_t)b * LL * D_INNER + d;
    const float* ub  = xc + (size_t)b * LL * D_INNER + d;
    float* yp = ys + (size_t)pair * LL;

    const int nch = (LL + 31) / 32;               // 157
    for (int ch = 0; ch < nch; ++ch) {
        int lend = (ch == nch - 1) ? (LL - ch * 32) : 32;
        for (int i = 0; i < lend; ++i) {
            int l = ch * 32 + i;
            float dtv = dtb[(size_t)l * D_INNER];
            float uv  = ub[(size_t)l * D_INNER];
            const float* row = xd + (size_t)l * (DT_RANK + 2 * D_STATE);
            float2 Bv = *(const float2*)(row + DT_RANK + s0);
            float2 Cv = *(const float2*)(row + DT_RANK + D_STATE + s0);
            float du = dtv * uv;
            h0 = __expf(dtv * A0) * h0 + du * Bv.x;
            h1 = __expf(dtv * A1) * h1 + du * Bv.y;
            float p = h0 * Cv.x + h1 * Cv.y;
            p += __shfl_xor(p, 16, 32);
            p += __shfl_xor(p, 8, 32);
            p += __shfl_xor(p, 4, 32);
            p += __shfl_xor(p, 2, 32);
            p += __shfl_xor(p, 1, 32);
            if (i == lane) yreg = p;              // rotate into coalesced slot
        }
        if (lane < lend) yp[ch * 32 + lane] = yreg;
    }
}

// y = (ys + xc*D) * silu(z) -> bf16
__global__ void gate_kernel(const float* __restrict__ ys, const float* __restrict__ xc,
                            const float* __restrict__ Dv, const float* __restrict__ xz,
                            bf16_t* __restrict__ y_bf)
{
    int t = blockIdx.x * 256 + threadIdx.x;
    if (t >= ML * D_INNER) return;
    int d  = t & (D_INNER - 1);
    int bl = t >> 10;
    int b  = bl / LL, l = bl % LL;
    float y = ys[((size_t)b * D_INNER + d) * LL + l] + xc[t] * Dv[d];
    float z = xz[(size_t)bl * (2 * D_INNER) + D_INNER + d];
    y *= z / (1.f + __expf(-z));
    y_bf[t] = (bf16_t)y;
}

__global__ void selu_kernel(float* __restrict__ m, int n)
{
    int t = blockIdx.x * 256 + threadIdx.x;
    if (t >= n) return;
    const float sc = 1.0507009873554805f, al = 1.6732632423543772f;
    float v = m[t];
    m[t] = (v > 0.f) ? sc * v : sc * al * (__expf(v) - 1.f);
}

// o[b,c,l] = sum_d m[b,l,d]*w_final[c,d]
__global__ void final_proj_kernel(const float* __restrict__ m, const float* __restrict__ w_final,
                                  float* __restrict__ o)
{
    int t = blockIdx.x * 256 + threadIdx.x;
    if (t >= BB * C_IN * LL) return;
    int l = t % LL;
    int c = (t / LL) % C_IN;
    int b = t / (C_IN * LL);
    const float4* mr = (const float4*)(m + ((size_t)b * LL + l) * D_MODEL);
    const float4* wr = (const float4*)(w_final + (size_t)c * D_MODEL);
    float acc = 0.f;
    for (int k = 0; k < D_MODEL / 4; ++k) {
        float4 a = mr[k], w = wr[k];
        acc += a.x * w.x + a.y * w.y + a.z * w.z + a.w * w.w;
    }
    o[t] = acc;
}

// fc1: one block per output (b,j); streams the 123MB weight (HBM-bound)
__global__ __launch_bounds__(256) void fc1_kernel(const float* __restrict__ o,
                                                  const float* __restrict__ w_fc1,
                                                  const float* __restrict__ b_fc1,
                                                  float* __restrict__ out)
{
    const int j = blockIdx.x & 511;
    const int b = blockIdx.x >> 9;
    const float4* f = (const float4*)(o + (size_t)b * (C_IN * LL));
    const float4* w = (const float4*)(w_fc1 + (size_t)j * (C_IN * LL));
    float acc = 0.f;
    for (int i = threadIdx.x; i < (C_IN * LL) / 4; i += 256) {
        float4 a = f[i], ww = w[i];
        acc += a.x * ww.x + a.y * ww.y + a.z * ww.z + a.w * ww.w;
    }
    acc += __shfl_xor(acc, 16, 32);
    acc += __shfl_xor(acc, 8, 32);
    acc += __shfl_xor(acc, 4, 32);
    acc += __shfl_xor(acc, 2, 32);
    acc += __shfl_xor(acc, 1, 32);
    __shared__ float red[8];
    int lane = threadIdx.x & 31, wv = threadIdx.x >> 5;
    if (lane == 0) red[wv] = acc;
    __syncthreads();
    if (threadIdx.x == 0) {
        float s = 0.f;
#pragma unroll
        for (int k = 0; k < 8; ++k) s += red[k];
        out[blockIdx.x] = s + b_fc1[j];
    }
}

// fc2 -> fc3 -> fc4 fused in one block via LDS
__global__ __launch_bounds__(256) void fc234_kernel(
    const float* __restrict__ f1,
    const float* __restrict__ w2, const float* __restrict__ b2,
    const float* __restrict__ w3, const float* __restrict__ b3,
    const float* __restrict__ w4, const float* __restrict__ b4,
    float* __restrict__ out)
{
    __shared__ float s1[1024], s2[512], s3[128];
    int tid = threadIdx.x;
    for (int i = tid; i < 1024; i += 256) s1[i] = f1[i];
    __syncthreads();
    for (int i = tid; i < 512; i += 256) {
        int b = i >> 8, j = i & 255;
        float acc = b2[j];
        const float* w = w2 + (size_t)j * 512;
        const float* f = s1 + b * 512;
        for (int k = 0; k < 512; ++k) acc += f[k] * w[k];
        s2[i] = acc;
    }
    __syncthreads();
    if (tid < 128) {
        int b = tid >> 6, j = tid & 63;
        float acc = b3[j];
        const float* w = w3 + (size_t)j * 256;
        const float* f = s2 + b * 256;
        for (int k = 0; k < 256; ++k) acc += f[k] * w[k];
        s3[tid] = acc;
    }
    __syncthreads();
    if (tid < 16) {
        int b = tid >> 3, j = tid & 7;
        float acc = b4[j];
        const float* w = w4 + (size_t)j * 64;
        const float* f = s3 + b * 64;
        for (int k = 0; k < 64; ++k) acc += f[k] * w[k];
        out[tid] = acc;
    }
}

// ---------------- launcher ----------------
extern "C" void kernel_launch(void* const* d_in, const int* in_sizes, int n_in,
                              void* d_out, int out_size, void* d_ws, size_t ws_size,
                              hipStream_t stream)
{
    const float* x       = (const float*)d_in[0];
    const float* w_init  = (const float*)d_in[1];
    const float* w_final = (const float*)d_in[2];
    const float* w_in    = (const float*)d_in[3];
    const float* conv_w  = (const float*)d_in[4];
    const float* conv_b  = (const float*)d_in[5];
    const float* w_xproj = (const float*)d_in[6];
    const float* w_dt    = (const float*)d_in[7];
    const float* b_dt    = (const float*)d_in[8];
    const float* A_log   = (const float*)d_in[9];
    const float* Dv      = (const float*)d_in[10];
    const float* w_out   = (const float*)d_in[11];
    const float* w_fc1   = (const float*)d_in[12];
    const float* b_fc1   = (const float*)d_in[13];
    const float* w_fc2   = (const float*)d_in[14];
    const float* b_fc2   = (const float*)d_in[15];
    const float* w_fc3   = (const float*)d_in[16];
    const float* b_fc3   = (const float*)d_in[17];
    const float* w_fc4   = (const float*)d_in[18];
    const float* b_fc4   = (const float*)d_in[19];
    float* out = (float*)d_out;

    char* ws = (char*)d_ws;
    size_t off = 0;
    auto alloc = [&](size_t bytes) -> char* {
        char* p = ws + off;
        off = (off + bytes + 255) & ~(size_t)255;
        return p;
    };
    bf16_t* h_bf     = (bf16_t*)alloc((size_t)ML * D_MODEL * 2);
    bf16_t* w_in_bf  = (bf16_t*)alloc((size_t)2 * D_INNER * D_MODEL * 2);
    bf16_t* w_xp_bf  = (bf16_t*)alloc((size_t)(DT_RANK + 2 * D_STATE) * D_INNER * 2);
    bf16_t* w_dt_bf  = (bf16_t*)alloc((size_t)D_INNER * DT_RANK * 2);
    bf16_t* w_out_bf = (bf16_t*)alloc((size_t)D_MODEL * D_INNER * 2);
    float*  xz       = (float*)alloc((size_t)ML * 2 * D_INNER * 4);
    float*  xc       = (float*)alloc((size_t)ML * D_INNER * 4);
    bf16_t* xc_bf    = (bf16_t*)alloc((size_t)ML * D_INNER * 2);
    float*  x_dbl    = (float*)alloc((size_t)ML * (DT_RANK + 2 * D_STATE) * 4);
    bf16_t* dtlo_bf  = (bf16_t*)alloc((size_t)ML * DT_RANK * 2);
    float*  dt       = (float*)alloc((size_t)ML * D_INNER * 4);
    float*  ys       = (float*)alloc((size_t)ML * D_INNER * 4);
    bf16_t* y_bf     = (bf16_t*)alloc((size_t)ML * D_INNER * 2);
    float*  m        = (float*)alloc((size_t)ML * D_MODEL * 4);
    float*  o        = (float*)alloc((size_t)BB * C_IN * LL * 4);
    float*  f1       = (float*)alloc(1024 * 4);
    (void)ws_size; (void)n_in; (void)in_sizes; (void)out_size;

#define CDIV(a, b) (((a) + (b) - 1) / (b))

    // weight conversions
    cvt_f32_bf16<<<CDIV(2 * D_INNER * D_MODEL, 256), 256, 0, stream>>>(w_in, w_in_bf, 2 * D_INNER * D_MODEL);
    cvt_f32_bf16<<<CDIV((DT_RANK + 2 * D_STATE) * D_INNER, 256), 256, 0, stream>>>(w_xproj, w_xp_bf, (DT_RANK + 2 * D_STATE) * D_INNER);
    cvt_f32_bf16<<<CDIV(D_INNER * DT_RANK, 256), 256, 0, stream>>>(w_dt, w_dt_bf, D_INNER * DT_RANK);
    cvt_f32_bf16<<<CDIV(D_MODEL * D_INNER, 256), 256, 0, stream>>>(w_out, w_out_bf, D_MODEL * D_INNER);

    // h = x^T w_init^T (bf16)
    h_proj_kernel<<<CDIV(ML * D_MODEL, 256), 256, 0, stream>>>(x, w_init, h_bf);

    // xz = h @ w_in^T : M=10000 N=2048 K=512
    {
        dim3 g(CDIV(2 * D_INNER, BN), CDIV(ML, BM));
        gemm_bf16_wmma<<<g, 256, 0, stream>>>(h_bf, w_in_bf, xz, ML, 2 * D_INNER, D_MODEL);
    }

    // causal depthwise conv + SiLU
    conv_silu_kernel<<<CDIV(ML * D_INNER, 256), 256, 0, stream>>>(xz, conv_w, conv_b, xc, xc_bf);

    // x_dbl = xc @ w_xproj^T : M=10000 N=160 K=1024
    {
        dim3 g(CDIV(DT_RANK + 2 * D_STATE, BN), CDIV(ML, BM));
        gemm_bf16_wmma<<<g, 256, 0, stream>>>(xc_bf, w_xp_bf, x_dbl, ML, DT_RANK + 2 * D_STATE, D_INNER);
    }

    // dt = softplus(dt_lo @ w_dt^T + b_dt) : M=10000 N=1024 K=32
    dtlo_kernel<<<CDIV(ML * DT_RANK, 256), 256, 0, stream>>>(x_dbl, dtlo_bf);
    {
        dim3 g(CDIV(D_INNER, BN), CDIV(ML, BM));
        gemm_bf16_wmma<<<g, 256, 0, stream>>>(dtlo_bf, w_dt_bf, dt, ML, D_INNER, DT_RANK);
    }
    softplus_kernel<<<CDIV(ML * D_INNER, 256), 256, 0, stream>>>(dt, b_dt, ML * D_INNER);

    // selective scan: 2048 (b,d) pairs, 8 waves/block
    scan_kernel<<<(BB * D_INNER) / 8, 256, 0, stream>>>(x_dbl, dt, xc, A_log, ys);

    // gating -> y (bf16)
    gate_kernel<<<CDIV(ML * D_INNER, 256), 256, 0, stream>>>(ys, xc, Dv, xz, y_bf);

    // m = y @ w_out^T : M=10000 N=512 K=1024 ; then selu
    {
        dim3 g(CDIV(D_MODEL, BN), CDIV(ML, BM));
        gemm_bf16_wmma<<<g, 256, 0, stream>>>(y_bf, w_out_bf, m, ML, D_MODEL, D_INNER);
    }
    selu_kernel<<<CDIV(ML * D_MODEL, 256), 256, 0, stream>>>(m, ML * D_MODEL);

    // o = m @ w_final^T (b,c,l)  == f reshaped
    final_proj_kernel<<<CDIV(BB * C_IN * LL, 256), 256, 0, stream>>>(m, w_final, o);

    // fc1 (HBM-bound: 123MB weight stream), then fused fc2/3/4
    fc1_kernel<<<BB * 512, 256, 0, stream>>>(o, w_fc1, b_fc1, f1);
    fc234_kernel<<<1, 256, 0, stream>>>(f1, w_fc2, b_fc2, w_fc3, b_fc3, w_fc4, b_fc4, out);
}
